// BipartiteLayer_29892972380779
// MI455X (gfx1250) — matrix operand order repaired and checked by reading, not compiled, not run on hardware
//
#include <hip/hip_runtime.h>

// ---------------------------------------------------------------------------
// Problem constants (from the reference)
// ---------------------------------------------------------------------------
#define N_NODES 100000
#define D_IN    128
#define D_OUT   64
#define A_HEADS 8
#define B_SEG   512
#define F_DIM   128        // 2*D_OUT
#define AGG_DIM 2048       // 2*F*A
#define H_DIM   2304       // D_IN + F + AGG_DIM
#define N_TILES 6250       // N_NODES / 16 (exact)

typedef __attribute__((ext_vector_type(16))) __bf16        v16bf;
typedef __attribute__((ext_vector_type(4)))  unsigned int  v4u;
typedef __attribute__((ext_vector_type(8)))  unsigned int  v8u;
typedef __attribute__((ext_vector_type(8)))  float         v8f;

__device__ __forceinline__ unsigned short f2bf(float f) {
  unsigned u = __float_as_uint(f);
  u += 0x7FFFu + ((u >> 16) & 1u);     // round-to-nearest-even
  return (unsigned short)(u >> 16);
}
__device__ __forceinline__ float bf2f(unsigned short h) {
  return __uint_as_float((unsigned)h << 16);
}

// Build a 16x32 bf16 A-fragment for one lane from a bf16 row (no VALU cvt).
// Per CDNA5 ISA (CDNA5 ISA, 16-bit A 16x32): lane L holds row M = L%16;
// elements e=0..7  -> K = k0..k0+7,  elements e=8..15 -> K = k0+16..k0+23,
// with k0 = kb + 8*laneHi.  Two contiguous 16-byte loads.
__device__ __forceinline__ v16bf load_afrag(const unsigned short* __restrict__ row, int k0) {
  const v4u lo = *(const v4u*)(row + k0);
  const v4u hi = *(const v4u*)(row + k0 + 16);
  v8u p;
  p[0] = lo[0]; p[1] = lo[1]; p[2] = lo[2]; p[3] = lo[3];
  p[4] = hi[0]; p[5] = hi[1]; p[6] = hi[2]; p[7] = hi[3];
  return __builtin_bit_cast(v16bf, p);
}

// ---------------------------------------------------------------------------
// Kernel 0a: fp32 -> bf16 for W_in [128,128] and W_out [64,2304]
// ---------------------------------------------------------------------------
__global__ void k_cvt_w(const float* __restrict__ Win,  const float* __restrict__ Wout,
                        unsigned short* __restrict__ Winb, unsigned short* __restrict__ Woutb) {
  int i = blockIdx.x * blockDim.x + threadIdx.x;
  if (i < F_DIM * D_IN)  Winb[i]  = f2bf(Win[i]);
  if (i < D_OUT * H_DIM) Woutb[i] = f2bf(Wout[i]);
}

// ---------------------------------------------------------------------------
// Kernel 0b: fp32 -> bf16 for x [N,128] (4 elements / thread)
// ---------------------------------------------------------------------------
__global__ void k_cvt_x(const float* __restrict__ x, unsigned short* __restrict__ xb) {
  const size_t i = ((size_t)blockIdx.x * blockDim.x + threadIdx.x) * 4;
  if (i >= (size_t)N_NODES * D_IN) return;
  const float4 v = *(const float4*)(x + i);
  ushort4 o;
  o.x = f2bf(v.x); o.y = f2bf(v.y); o.z = f2bf(v.z); o.w = f2bf(v.w);
  *(ushort4*)(xb + i) = o;
}

// ---------------------------------------------------------------------------
// Kernel 1: xp = x @ W_in^T + b_in   via v_wmma_f32_16x16x32_bf16
// one wave per 16-node tile; 8 column tiles (F=128), K=128 in 4 steps.
// Output written directly as bf16 (consumed by k_score / k_seg / k_out).
// ---------------------------------------------------------------------------
__global__ void k_xp(const unsigned short* __restrict__ xb,
                     const unsigned short* __restrict__ Winb,
                     const float* __restrict__ b_in,
                     unsigned short* __restrict__ xpb) {
  const int wave = blockIdx.x * (blockDim.x >> 5) + (threadIdx.x >> 5);
  if (wave >= N_TILES) return;                 // wave-uniform: EXEC all-1s in WMMA
  const int lane = threadIdx.x & 31;
  const int m    = lane & 15;
  const int hi   = lane >> 4;

  const unsigned short* rowA = xb + (size_t)(wave * 16 + m) * D_IN;
  v8f acc[8] = {};

#pragma unroll
  for (int ks = 0; ks < 4; ++ks) {
    const int kb = ks * 32;
    const v16bf a = load_afrag(rowA, kb + hi * 8);
#pragma unroll
    for (int nt = 0; nt < 8; ++nt) {
      // B 32x16 layout: lane's column N = lane%16; same K order as A:
      // e=0..7 -> K k0..k0+7, e=8..15 -> K k0+16..k0+23, k0 = kb + 8*laneHi.
      const v16bf b = load_afrag(Winb + (size_t)(nt * 16 + m) * D_IN, kb + hi * 8);
      acc[nt] = __builtin_amdgcn_wmma_f32_16x16x32_bf16(
          false, a, false, b, (short)0, acc[nt], false, false);
    }
  }

#pragma unroll
  for (int nt = 0; nt < 8; ++nt) {
    const int col  = nt * 16 + m;
    const float bs = b_in[col];
#pragma unroll
    for (int r = 0; r < 8; ++r) {
      const int row = wave * 16 + r + hi * 8;          // C-layout: VGPR r -> M
      xpb[(size_t)row * F_DIM + col] = f2bf(acc[nt][r] + bs);
    }
  }
}

// ---------------------------------------------------------------------------
// Kernel 2: score = exp(-|xp @ W_a^T + b_a|)   (scalar per node; W_a in LDS)
// ---------------------------------------------------------------------------
__global__ void k_score(const unsigned short* __restrict__ xpb,
                        const float* __restrict__ W_a,
                        const float* __restrict__ b_a,
                        float* __restrict__ score) {
  __shared__ float sWa[A_HEADS * F_DIM];
  for (int i = threadIdx.x; i < A_HEADS * F_DIM; i += blockDim.x) sWa[i] = W_a[i];
  __syncthreads();
  const int n = blockIdx.x * blockDim.x + threadIdx.x;
  if (n >= N_NODES) return;
  const unsigned short* row = xpb + (size_t)n * F_DIM;
  float s[A_HEADS];
#pragma unroll
  for (int a = 0; a < A_HEADS; ++a) s[a] = b_a[a];
  for (int f8 = 0; f8 < F_DIM; f8 += 8) {
    const v4u ld = *(const v4u*)(row + f8);             // 8 bf16
#pragma unroll
    for (int j = 0; j < 4; ++j) {
      const float x0 = __uint_as_float(ld[j] << 16);
      const float x1 = __uint_as_float(ld[j] & 0xFFFF0000u);
#pragma unroll
      for (int a = 0; a < A_HEADS; ++a) {
        s[a] = fmaf(x0, sWa[a * F_DIM + f8 + 2 * j],     s[a]);
        s[a] = fmaf(x1, sWa[a * F_DIM + f8 + 2 * j + 1], s[a]);
      }
    }
  }
#pragma unroll
  for (int a = 0; a < A_HEADS; ++a)
    score[(size_t)n * A_HEADS + a] = __expf(-fabsf(s[a]));
}

// ---------------------------------------------------------------------------
// Kernel 3: segment mean/max -> agg_bf [B, 2*F*A] (bf16; only the GEMM reads it)
// batch is sorted: one workgroup per segment, binary-searched bounds, no atomics.
// Thread t owns 4 consecutive (a,f) cells out of A*F=1024.
// ---------------------------------------------------------------------------
__device__ __forceinline__ int lbound(const int* __restrict__ arr, int n, int key) {
  int lo = 0, hi = n;
  while (lo < hi) { int mid = (lo + hi) >> 1; if (arr[mid] < key) lo = mid + 1; else hi = mid; }
  return lo;
}

__global__ void k_seg(const unsigned short* __restrict__ xpb,
                      const float* __restrict__ score,
                      const int* __restrict__ batch,
                      unsigned short* __restrict__ aggb) {
  const int b     = blockIdx.x;                       // 0..511
  const int start = lbound(batch, N_NODES, b);
  const int end   = lbound(batch, N_NODES, b + 1);

  const int j0 = threadIdx.x * 4;                     // 0..1020
  const int a  = j0 >> 7;                             // head (4 cells share it)
  const int f0 = j0 & 127;

  float sum[4] = {0.f, 0.f, 0.f, 0.f};
  float mx[4]  = {-INFINITY, -INFINITY, -INFINITY, -INFINITY};
  for (int n = start; n < end; ++n) {
    const float sc = score[(size_t)n * A_HEADS + a];            // wave-uniform addr
    const uint2 xv = *(const uint2*)(xpb + (size_t)n * F_DIM + f0);   // 4 bf16
    const float e0 = sc * __uint_as_float(xv.x << 16);
    const float e1 = sc * __uint_as_float(xv.x & 0xFFFF0000u);
    const float e2 = sc * __uint_as_float(xv.y << 16);
    const float e3 = sc * __uint_as_float(xv.y & 0xFFFF0000u);
    sum[0] += e0; sum[1] += e1; sum[2] += e2; sum[3] += e3;
    mx[0] = fmaxf(mx[0], e0); mx[1] = fmaxf(mx[1], e1);
    mx[2] = fmaxf(mx[2], e2); mx[3] = fmaxf(mx[3], e3);
  }
  const int   cnt = end - start;
  const float inv = 1.0f / fmaxf((float)cnt, 1.0f);
  // agg row layout: [A][2F]: mean at a*2F + f, max at a*2F + F + f
  const size_t base = (size_t)b * AGG_DIM + (size_t)a * (2 * F_DIM) + f0;
#pragma unroll
  for (int i = 0; i < 4; ++i) {
    aggb[base + i]         = f2bf(sum[i] * inv);
    aggb[base + F_DIM + i] = f2bf((cnt > 0) ? mx[i] : 0.f);
  }
}

// ---------------------------------------------------------------------------
// Kernel 4: out = relu([x | xp | agg[batch]] @ W_out^T + b_out)
// one wave per 16-node tile; 4 accumulators (64 cols); K = 2304 in 72 steps.
// All A/B operands pre-converted bf16 -> inner loop is pure load+WMMA.
// ---------------------------------------------------------------------------
__global__ void k_out(const unsigned short* __restrict__ xb,
                      const unsigned short* __restrict__ xpb,
                      const unsigned short* __restrict__ aggb,
                      const int* __restrict__ batch,
                      const unsigned short* __restrict__ Woutb,
                      const float* __restrict__ b_out,
                      float* __restrict__ out) {
  const int wave = blockIdx.x * (blockDim.x >> 5) + (threadIdx.x >> 5);
  if (wave >= N_TILES) return;
  const int lane = threadIdx.x & 31;
  const int m    = lane & 15;
  const int hi   = lane >> 4;

  const int nm   = wave * 16 + m;               // A-row owned by this lane
  const int bseg = batch[nm];
  const unsigned short* srcs[3];
  srcs[0] = xb   + (size_t)nm * D_IN;
  srcs[1] = xpb  + (size_t)nm * F_DIM;
  srcs[2] = aggb + (size_t)bseg * AGG_DIM;      // gathered; batch sorted -> L2-friendly
  const int klen[3] = {D_IN, F_DIM, AGG_DIM};

  v8f acc[4] = {};
  int wbase = 0;
  for (int s = 0; s < 3; ++s) {
    const unsigned short* row = srcs[s];
    for (int kb = 0; kb < klen[s]; kb += 32) {
      const v16bf a = load_afrag(row, kb + hi * 8);
#pragma unroll
      for (int nt = 0; nt < 4; ++nt) {
        const v16bf b = load_afrag(Woutb + (size_t)(nt * 16 + m) * H_DIM
                                   + wbase, kb + hi * 8);
        acc[nt] = __builtin_amdgcn_wmma_f32_16x16x32_bf16(
            false, a, false, b, (short)0, acc[nt], false, false);
      }
    }
    wbase += klen[s];
  }

#pragma unroll
  for (int nt = 0; nt < 4; ++nt) {
    const int   col = nt * 16 + m;
    const float bs  = b_out[col];
#pragma unroll
    for (int r = 0; r < 8; ++r) {
      const int   node = wave * 16 + r + hi * 8;
      const float v    = acc[nt][r] + bs;
      out[(size_t)node * D_OUT + col] = v > 0.f ? v : 0.f;
    }
  }
}

// ---------------------------------------------------------------------------
// Launcher. Inputs (setup_inputs order):
//  0:x f32[N,128] 1:batch i32[N] 2:num_segments i32[1] 3:W_in f32[128,128]
//  4:b_in f32[128] 5:W_a f32[8,128] 6:b_a f32[8] 7:W_out f32[64,2304] 8:b_out f32[64]
// Workspace (all 32B aligned, ~57 MB total):
//  x_bf u16[N*128] | xp_bf u16[N*128] | score f32[N*8] | agg_bf u16[512*2048]
//  | Winb u16[16384] | Woutb u16[147456]
// ---------------------------------------------------------------------------
extern "C" void kernel_launch(void* const* d_in, const int* in_sizes, int n_in,
                              void* d_out, int out_size, void* d_ws, size_t ws_size,
                              hipStream_t stream) {
  const float* x     = (const float*)d_in[0];
  const int*   batch = (const int*)d_in[1];
  const float* W_in  = (const float*)d_in[3];
  const float* b_in  = (const float*)d_in[4];
  const float* W_a   = (const float*)d_in[5];
  const float* b_a   = (const float*)d_in[6];
  const float* W_out = (const float*)d_in[7];
  const float* b_out = (const float*)d_in[8];
  float*       out   = (float*)d_out;

  char* ws = (char*)d_ws;
  unsigned short* xb    = (unsigned short*)ws;  ws += (size_t)N_NODES * D_IN  * 2;
  unsigned short* xpb   = (unsigned short*)ws;  ws += (size_t)N_NODES * F_DIM * 2;
  float*          score = (float*)ws;           ws += (size_t)N_NODES * A_HEADS * 4;
  unsigned short* aggb  = (unsigned short*)ws;  ws += (size_t)B_SEG * AGG_DIM * 2;
  unsigned short* Winb  = (unsigned short*)ws;  ws += (size_t)F_DIM * D_IN * 2;
  unsigned short* Woutb = (unsigned short*)ws;

  // 0) one-shot bf16 conversions
  k_cvt_w<<<(D_OUT * H_DIM + 255) / 256, 256, 0, stream>>>(W_in, W_out, Winb, Woutb);
  k_cvt_x<<<((N_NODES * D_IN / 4) + 255) / 256, 256, 0, stream>>>(x, xb);
  // 1) xp GEMM (WMMA bf16, bf16 output)
  k_xp<<<(N_TILES + 7) / 8, 256, 0, stream>>>(xb, Winb, b_in, xpb);
  // 2) scores
  k_score<<<(N_NODES + 255) / 256, 256, 0, stream>>>(xpb, W_a, b_a, score);
  // 3) segment mean/max (one workgroup per segment, bf16 output)
  k_seg<<<B_SEG, 256, 0, stream>>>(xpb, score, batch, aggb);
  // 4) fused gather + big GEMM (WMMA bf16) + bias + ReLU
  k_out<<<(N_TILES + 7) / 8, 256, 0, stream>>>(xb, xpb, aggb, batch, Woutb, b_out, out);
}
